// TaskGraphGAT_4157528343020
// MI455X (gfx1250) — matrix operand order repaired and checked by reading, not compiled, not run on hardware
//
#include <hip/hip_runtime.h>
#include <hip/hip_bf16.h>

typedef __bf16 bf16_t;
typedef bf16_t bf16x8  __attribute__((ext_vector_type(8)));
typedef bf16_t bf16x16 __attribute__((ext_vector_type(16)));
typedef float  f32x8   __attribute__((ext_vector_type(8)));

#define HID   128
#define NTASK 8
#define NHEAD 4
#define BT    8              // batch items per block
#define ROWS  (BT * NTASK)   // 64 rows per block
#define WHS   520            // padded stride (bf16 elems) of activation buffer
#define WB1   136            // padded stride for GEMM1 weight stage
#define WB2   520            // padded stride for GEMM2 weight stage

static __device__ __forceinline__ f32x8 wmma_bf16(bf16x16 a, bf16x16 b, f32x8 c) {
  return __builtin_amdgcn_wmma_f32_16x16x32_bf16(false, a, false, b, (short)0, c, false, false);
}

// A fragment (16-bit, 16x32): lane holds row M=lane&15; K chunks [koff..koff+7], [koff+16..koff+23]
static __device__ __forceinline__ bf16x16 ld_afrag(const bf16_t* p) {
  bf16x16 v;
  ((bf16x8*)&v)[0] = *(const bf16x8*)(p);
  ((bf16x8*)&v)[1] = *(const bf16x8*)(p + 16);
  return v;
}
// B fragment (16-bit, 32x16): lane holds col N=lane&15; K chunks [koff..koff+7], [koff+16..koff+23]
static __device__ __forceinline__ bf16x16 ld_bfrag(const bf16_t* p) {
  bf16x16 v;
  ((bf16x8*)&v)[0] = *(const bf16x8*)(p);
  ((bf16x8*)&v)[1] = *(const bf16x8*)(p + 16);
  return v;
}

// ---------- prep: h_task = task_emb @ W_in^T + b_in ; ew ----------
__global__ void gat_prep_small(const float* __restrict__ temb, const float* __restrict__ Win,
                               const float* __restrict__ bin, const int* __restrict__ adj,
                               const int* __restrict__ et, const float* __restrict__ omi,
                               const float* __restrict__ omc, float* __restrict__ htask,
                               float* __restrict__ ew_out) {
  int t = threadIdx.x;  // 128 threads
  for (int idx = t; idx < NTASK * HID; idx += 128) {
    int n = idx >> 7, g = idx & 127;
    float s = bin[g];
    for (int k = 0; k < 64; ++k) s += temb[n * 64 + k] * Win[g * 64 + k];
    htask[idx] = s;
  }
  if (t < 64) {
    float w = 1.f;
    int e = et[t];
    if (e == 1) w = omi[0]; else if (e == 2) w = omc[0];
    ew_out[t] = w * (float)adj[t];
  }
}

// ---------- prep: weights f32 -> bf16 (both layers; each array is 131072 elems) ----------
__global__ void gat_prep_cvt(const float* __restrict__ Wh, const float* __restrict__ oW,
                             bf16_t* __restrict__ wb, bf16_t* __restrict__ ob) {
  int i = blockIdx.x * blockDim.x + threadIdx.x;
  if (i < 131072) { wb[i] = (bf16_t)Wh[i]; ob[i] = (bf16_t)oW[i]; }
}

// ---------- fused GAT layer ----------
__global__ __launch_bounds__(128)
void gat_layer_kernel(const float* __restrict__ h_prev,    // null for layer 0
                      const float* __restrict__ shared_emb,
                      const float* __restrict__ h_task,
                      const bf16_t* __restrict__ Wb,       // [4][128][128] bf16
                      const float* __restrict__ a_h,       // [4][256]
                      const bf16_t* __restrict__ oW,       // [128][512] bf16
                      const float* __restrict__ out_b,
                      const float* __restrict__ ln_s,
                      const float* __restrict__ ln_b,
                      const int* __restrict__ adj,
                      const int* __restrict__ et,
                      const float* __restrict__ omi,
                      const float* __restrict__ omc,
                      float* __restrict__ h_out) {
  __shared__ __align__(16) bf16_t Whs[ROWS * WHS];      // 66,560 B  per-head activations
  __shared__ __align__(16) bf16_t wbuf[64 * WB2];       // 66,560 B  staged weights
  __shared__ float attS[BT][NHEAD][8][8];               //  8,192 B
  __shared__ float eiS[ROWS][NHEAD];
  __shared__ float ejS[ROWS][NHEAD];
  __shared__ float ewS[64];
  __shared__ int   adjS[64];

  const int tid  = threadIdx.x;
  const int w    = tid >> 5;
  const int lane = tid & 31;
  const int l15  = lane & 15;
  const int hi   = (lane >> 4) & 1;
  const int koff = hi * 8;
  const int rowbase = blockIdx.x * ROWS;

  if (tid < 64) {
    float wg = 1.f;
    int e = et[tid];
    if (e == 1) wg = omi[0]; else if (e == 2) wg = omc[0];
    ewS[tid]  = wg * (float)adj[tid];
    adjS[tid] = adj[tid];
  }

  // ---- build GEMM1 A fragments in registers ----
  const int arow = w * 16 + l15;
  const int grow = rowbase + arow;
  const int gb = grow >> 3, gn = grow & 7;
  bf16x16 afrag[4];
  {
    const float* hrow = h_prev ? (h_prev + (size_t)grow * HID) : nullptr;
    const float* srow = shared_emb + (size_t)gb * HID;
    const float* trow = h_task + gn * HID;
    #pragma unroll
    for (int kb = 0; kb < 4; ++kb) {
      int k0 = kb * 32 + koff;
      #pragma unroll
      for (int j = 0; j < 16; ++j) {
        int k = k0 + (j < 8 ? j : 8 + j);  // chunks [k0..k0+7], [k0+16..k0+23]
        float v = hrow ? hrow[k] : (srow[k] + trow[k]);
        afrag[kb][j] = (bf16_t)v;
      }
    }
  }

  // ---- GEMM1: per-head projection, Whs[row][h*128+f] = (h @ W_h^T) ----
  for (int h = 0; h < NHEAD; ++h) {
    __syncthreads();  // wbuf reuse guard
    const bf16_t* wsrc = Wb + h * (HID * HID);
    if (h < NHEAD - 1) __builtin_prefetch(Wb + (h + 1) * (HID * HID), 0, 0);
    for (int i = tid; i < 2048; i += 128) {               // 2048 x 16B
      int c = i >> 4, k = (i & 15) * 8;
      *(bf16x8*)&wbuf[c * WB1 + k] = *(const bf16x8*)&wsrc[c * HID + k];
    }
    __syncthreads();
    f32x8 acc[8];
    #pragma unroll
    for (int ct = 0; ct < 8; ++ct) acc[ct] = {};
    #pragma unroll
    for (int kb = 0; kb < 4; ++kb) {
      #pragma unroll
      for (int ct = 0; ct < 8; ++ct) {
        bf16x16 bfrag = ld_bfrag(&wbuf[(ct * 16 + l15) * WB1 + kb * 32 + hi * 8]);
        acc[ct] = wmma_bf16(afrag[kb], bfrag, acc[ct]);
      }
    }
    #pragma unroll
    for (int ct = 0; ct < 8; ++ct) {
      int col = h * HID + ct * 16 + l15;
      #pragma unroll
      for (int v = 0; v < 8; ++v) {
        int rl = w * 16 + v + hi * 8;     // C layout: VGPR v -> row v (+8 upper half)
        Whs[rl * WHS + col] = (bf16_t)acc[ct][v];
      }
    }
  }
  __syncthreads();

  // ---- ei/ej: dot(Wh, a_src/a_dst) ----
  for (int p = tid; p < ROWS * NHEAD; p += 128) {
    int row = p >> 2, h = p & 3;
    const float* as = a_h + h * 256;
    const bf16_t* wr = &Whs[row * WHS + h * HID];
    float si = 0.f, sj = 0.f;
    for (int k8 = 0; k8 < 16; ++k8) {
      bf16x8 v = *(const bf16x8*)(wr + k8 * 8);
      #pragma unroll
      for (int j = 0; j < 8; ++j) {
        float x = (float)v[j];
        si += x * as[k8 * 8 + j];
        sj += x * as[128 + k8 * 8 + j];
      }
    }
    eiS[row][h] = si; ejS[row][h] = sj;
  }
  __syncthreads();

  // ---- attention softmax (faithful: masked -> -9e15 * 0 = 0 logit) ----
  if (tid < BT * NHEAD) {
    int bl = tid >> 2, h = tid & 3;
    int rb = bl * 8;
    #pragma unroll
    for (int n = 0; n < 8; ++n) {
      float ein = eiS[rb + n][h];
      float lg[8], mx = -3.4e38f;
      #pragma unroll
      for (int m = 0; m < 8; ++m) {
        float x = ein + ejS[rb + m][h];
        float e = x > 0.f ? x : 0.2f * x;                   // leaky_relu 0.2
        float L = (adjS[n * 8 + m] ? e : -9e15f) * ewS[n * 8 + m];
        lg[m] = L; mx = fmaxf(mx, L);
      }
      float s = 0.f;
      #pragma unroll
      for (int m = 0; m < 8; ++m) { lg[m] = __expf(lg[m] - mx); s += lg[m]; }
      float inv = 1.f / s;
      #pragma unroll
      for (int m = 0; m < 8; ++m) attS[bl][h][n][m] = lg[m] * inv;
    }
  }
  __syncthreads();

  // ---- hp = elu(att @ Wh), in place over Whs (chunked, barrier-protected) ----
  {
    const int r = tid >> 1;
    const int bl2 = r >> 3, n2 = r & 7, rb2 = bl2 * 8;
    for (int ch = 0; ch < 8; ++ch) {
      int h = ch >> 1;
      int col0 = ch * 64 + (tid & 1) * 32;
      float av[8];
      #pragma unroll
      for (int m = 0; m < 8; ++m) av[m] = attS[bl2][h][n2][m];
      float acc[32];
      #pragma unroll
      for (int j = 0; j < 32; ++j) acc[j] = 0.f;
      #pragma unroll
      for (int m = 0; m < 8; ++m) {
        const bf16_t* src = &Whs[(rb2 + m) * WHS + col0];
        #pragma unroll
        for (int q = 0; q < 4; ++q) {
          bf16x8 v = *(const bf16x8*)(src + q * 8);
          #pragma unroll
          for (int j = 0; j < 8; ++j) acc[q * 8 + j] += av[m] * (float)v[j];
        }
      }
      bf16x8 outv[4];
      #pragma unroll
      for (int q = 0; q < 4; ++q)
        #pragma unroll
        for (int j = 0; j < 8; ++j) {
          float x = acc[q * 8 + j];
          outv[q][j] = (bf16_t)(x > 0.f ? x : (__expf(x) - 1.f));   // elu
        }
      __syncthreads();
      bf16_t* dst = &Whs[r * WHS + col0];
      #pragma unroll
      for (int q = 0; q < 4; ++q) *(bf16x8*)(dst + q * 8) = outv[q];
      __syncthreads();
    }
  }

  // ---- GEMM2: gout = hcat @ out_W^T, streamed B in 2 x 64-col chunks ----
  f32x8 acc2[8];
  #pragma unroll
  for (int g = 0; g < 8; ++g) acc2[g] = {};
  #pragma unroll
  for (int chunk = 0; chunk < 2; ++chunk) {
    __syncthreads();
    for (int i = tid; i < 4096; i += 128) {               // 64 cols x 512 K, 16B units
      int c = i >> 6, k = (i & 63) * 8;
      *(bf16x8*)&wbuf[c * WB2 + k] = *(const bf16x8*)&oW[(chunk * 64 + c) * 512 + k];
    }
    __syncthreads();
    for (int kb = 0; kb < 16; ++kb) {
      bf16x16 af2 = ld_afrag(&Whs[(w * 16 + l15) * WHS + kb * 32 + koff]);
      #pragma unroll
      for (int ctl = 0; ctl < 4; ++ctl) {
        bf16x16 bf = ld_bfrag(&wbuf[(ctl * 16 + l15) * WB2 + kb * 32 + hi * 8]);
        acc2[chunk * 4 + ctl] = wmma_bf16(af2, bf, acc2[chunk * 4 + ctl]);
      }
    }
  }

  // ---- epilogue: +bias, residual, layernorm (row stats via shfl_xor in 16-lane halves) ----
  float obv[8], lsv[8], lbv[8];
  #pragma unroll
  for (int g = 0; g < 8; ++g) {
    int col = g * 16 + l15;
    obv[g] = out_b[col]; lsv[g] = ln_s[col]; lbv[g] = ln_b[col];
  }
  float xs[8][8];  // [g][v]
  #pragma unroll
  for (int v = 0; v < 8; ++v) {
    int rl = w * 16 + v + hi * 8;
    int gr = rowbase + rl;
    int b = gr >> 3, n = gr & 7;
    const float* hrow = h_prev ? (h_prev + (size_t)gr * HID) : nullptr;
    const float* srow = shared_emb + (size_t)b * HID;
    const float* trow = h_task + n * HID;
    #pragma unroll
    for (int g = 0; g < 8; ++g) {
      int col = g * 16 + l15;
      float hv = hrow ? hrow[col] : (srow[col] + trow[col]);
      xs[g][v] = hv + acc2[g][v] + obv[g];
    }
  }
  #pragma unroll
  for (int v = 0; v < 8; ++v) {
    float s = 0.f, q = 0.f;
    #pragma unroll
    for (int g = 0; g < 8; ++g) { s += xs[g][v]; q += xs[g][v] * xs[g][v]; }
    #pragma unroll
    for (int m = 1; m < 16; m <<= 1) { s += __shfl_xor(s, m, 32); q += __shfl_xor(q, m, 32); }
    float mean = s * (1.f / 128.f);
    float var  = q * (1.f / 128.f) - mean * mean;
    float rstd = rsqrtf(var + 1e-5f);
    int gr = rowbase + w * 16 + v + hi * 8;
    float* orow = h_out + (size_t)gr * HID;
    #pragma unroll
    for (int g = 0; g < 8; ++g) {
      int col = g * 16 + l15;
      orow[col] = (xs[g][v] - mean) * rstd * lsv[g] + lbv[g];
    }
  }
}

extern "C" void kernel_launch(void* const* d_in, const int* in_sizes, int n_in,
                              void* d_out, int out_size, void* d_ws, size_t ws_size,
                              hipStream_t stream) {
  const float* shared_emb = (const float*)d_in[0];
  const float* task_emb   = (const float*)d_in[1];
  const float* W_in       = (const float*)d_in[2];
  const float* b_in       = (const float*)d_in[3];
  const float* W_heads    = (const float*)d_in[4];   // [2][4][128][128]
  const float* a_heads    = (const float*)d_in[5];   // [2][4][256]
  const float* out_W      = (const float*)d_in[6];   // [2][128][512]
  const float* out_b      = (const float*)d_in[7];
  const float* ln_scale   = (const float*)d_in[8];
  const float* ln_bias    = (const float*)d_in[9];
  const float* omi        = (const float*)d_in[10];
  const float* omc        = (const float*)d_in[11];
  const int*   adj        = (const int*)d_in[12];
  const int*   et         = (const int*)d_in[13];

  const int B = in_sizes[0] / HID;
  float* h_out  = (float*)d_out;
  float* ew_out = h_out + (size_t)B * NTASK * HID;

  float*  ws_htask = (float*)d_ws;                           // 4 KB
  bf16_t* ws_wb = (bf16_t*)((char*)d_ws + 4096);             // 2*4*128*128 bf16
  bf16_t* ws_ow = ws_wb + 2 * NHEAD * HID * HID;             // 2*128*512 bf16

  gat_prep_small<<<1, 128, 0, stream>>>(task_emb, W_in, b_in, adj, et, omi, omc,
                                        ws_htask, ew_out);
  gat_prep_cvt<<<512, 256, 0, stream>>>(W_heads, out_W, ws_wb, ws_ow);

  const int nblocks = B / BT;
  // layer 0 (input = h_task + shared), output -> d_out h region
  gat_layer_kernel<<<nblocks, 128, 0, stream>>>(
      nullptr, shared_emb, ws_htask,
      ws_wb, a_heads, ws_ow, out_b, ln_scale, ln_bias,
      adj, et, omi, omc, h_out);
  // layer 1 (input = layer-0 output, in place)
  gat_layer_kernel<<<nblocks, 128, 0, stream>>>(
      h_out, shared_emb, ws_htask,
      ws_wb + NHEAD * HID * HID, a_heads + NHEAD * 256, ws_ow + HID * 512,
      out_b + HID, ln_scale + HID, ln_bias + HID,
      adj, et, omi, omc, h_out);
}
